// CIG_SA_attention_29326036697209
// MI455X (gfx1250) — matrix operand order repaired and checked by reading, not compiled, not run on hardware
//
#include <hip/hip_runtime.h>
#include <hip/hip_bf16.h>
#include <math.h>

typedef __bf16 bf16t;
typedef __attribute__((ext_vector_type(4)))  __bf16 v4bf;
typedef __attribute__((ext_vector_type(16))) __bf16 v16bf;
typedef __attribute__((ext_vector_type(8)))  float  v8f;

#define CB     16
#define CC     128
#define CH     128
#define CWD    128
#define CHW    (CH * CWD)     // 16384
#define SPITCH 132            // padded f32 row pitch for softmax scratch
#define BP     136            // padded bf16 pitch for transposed tiles (136*2 % 16 == 0)
#define CCHUNK 4              // channels per attention block

__device__ __forceinline__ float sigmoidf_(float z) {
    // v_rcp_f32 + v_exp_f32 instead of the IEEE div expansion
    return __builtin_amdgcn_rcpf(1.0f + __expf(-z));
}

// ---- WMMA fragment loaders (bf16, 16x16x32, per CDNA5 ISA layout tables) ----

// A tile 16(M)x32(K) bf16 from LDS, row-major with row stride ld (elements).
// lane<16: e0..7 -> K=0..7, e8..15 -> K=16..23 ; lane>=16: +8 on K groups.
__device__ __forceinline__ v16bf load_A16x32(const bf16t* base, int ld) {
    const int lane = threadIdx.x & 31;
    const int half = lane >> 4, m = lane & 15;
    const bf16t* p = base + m * ld + (half << 3);
    union { v16bf v; struct { float4 lo, hi; } q; } u;
    u.q.lo = *reinterpret_cast<const float4*>(p);        // K group 0 (8 bf16)
    u.q.hi = *reinterpret_cast<const float4*>(p + 16);   // K group 1 (8 bf16)
    return u.v;
}

// B tile 32(K)x16(N), element (k,n) at base[k + n*sn]  (k contiguous).
// lane<16: e0..7 -> K=0..7, e8..15 -> K=16..23 ; lane>=16: +8 on K groups ; col = lane&15.
// sn must be a multiple of 8 (16-byte alignment of the b128 loads).
__device__ __forceinline__ v16bf load_B_k1(const bf16t* base, int sn) {
    const int lane = threadIdx.x & 31;
    const int half = lane >> 4, n = lane & 15;
    const bf16t* p = base + n * sn + (half << 3);
    union { v16bf v; struct { float4 a, b; } q; } u;
    u.q.a = *reinterpret_cast<const float4*>(p);
    u.q.b = *reinterpret_cast<const float4*>(p + 16);
    return u.v;
}

// ---------------- Kernel 0: du_logit <- b_du ----------------
__global__ void k_init_du(float* __restrict__ du, const float* __restrict__ b_du) {
    int i = blockIdx.x * 256 + threadIdx.x;
    if (i < CB * CHW) du[i] = b_du[0];
}

// ---------------- Kernel 1: gate GEMM + y + channel pooling -> feat ----------------
// One block per (b, h): all 128 channels x 128 w of one image row.
__global__ __launch_bounds__(256) void k_gate_feat(
    const float* __restrict__ x, const float* __restrict__ w_cw,
    const float* __restrict__ b_cw, float* __restrict__ feat)
{
    __shared__ bf16t wlds[CC * CC];     // 32 KB  W_cw bf16, [o][i]
    __shared__ bf16t xbT[CWD * BP];     // 34 KB  x row bf16 TRANSPOSED: [w][i]
    __shared__ float xf[CC * CWD];      // 64 KB  x row f32, [i][w]
    __shared__ float psum[8][CWD];      // per-wave partial sum(y)
    __shared__ float pmax[8][CWD];      // per-wave partial max(y)
    __shared__ float bc[CC];

    const int tid = threadIdx.x;
    const int img = blockIdx.x >> 7;
    const int row = blockIdx.x & 127;

    const float* xrow = x + (size_t)img * CC * CHW + (size_t)row * CWD;
#pragma unroll
    for (int it = 0; it < 16; ++it) {
        int l4 = (tid + it * 256) * 4;            // l4 = ci*128 + p, p multiple of 4
        int ci = l4 >> 7, p = l4 & 127;

        float4 wv4 = *reinterpret_cast<const float4*>(w_cw + l4);
        v4bf wp;
        wp[0] = (bf16t)wv4.x; wp[1] = (bf16t)wv4.y;
        wp[2] = (bf16t)wv4.z; wp[3] = (bf16t)wv4.w;
        *reinterpret_cast<v4bf*>(&wlds[l4]) = wp;

        float4 xv4 = *reinterpret_cast<const float4*>(xrow + (size_t)ci * CHW + p);
        *reinterpret_cast<float4*>(&xf[l4]) = xv4;
        xbT[(p + 0) * BP + ci] = (bf16t)xv4.x;    // transposed: K(=ci) contiguous
        xbT[(p + 1) * BP + ci] = (bf16t)xv4.y;
        xbT[(p + 2) * BP + ci] = (bf16t)xv4.z;
        xbT[(p + 3) * BP + ci] = (bf16t)xv4.w;
    }
    if (tid < CC) bc[tid] = b_cw[tid];
    __syncthreads();

    const int wid  = tid >> 5, lane = tid & 31;
    const int half = lane >> 4, nn = lane & 15;

    // Each wave owns output channels [16*wid, 16*wid+16); hoist A across pixel tiles.
    v16bf Atile[4];
#pragma unroll
    for (int kt = 0; kt < 4; ++kt)
        Atile[kt] = load_A16x32(wlds + (wid * 16) * CC + kt * 32, CC);

#pragma unroll
    for (int pt = 0; pt < 8; ++pt) {
        v8f acc = {};
#pragma unroll
        for (int kt = 0; kt < 4; ++kt) {
            // B[k=i][n=p] = xbT[p*BP + i]: k contiguous
            v16bf Bt = load_B_k1(xbT + (pt * 16) * BP + kt * 32, BP);
            acc = __builtin_amdgcn_wmma_f32_16x16x32_bf16(
                false, Atile[kt], false, Bt, (short)0, acc, false, false);
        }
        const int p = pt * 16 + nn;
        float sy = 0.0f, my = -3.0e38f;
#pragma unroll
        for (int e = 0; e < 8; ++e) {
            int o = wid * 16 + half * 8 + e;
            float g = acc[e] + bc[o];
            float gate = sigmoidf_(fmaxf(g, 0.0f));
            float yv = xf[o * CWD + p] * gate;
            sy += yv;
            my = fmaxf(my, yv);
        }
        sy += __shfl_xor(sy, 16);
        my  = fmaxf(my, __shfl_xor(my, 16));
        if (half == 0) { psum[wid][p] = sy; pmax[wid][p] = my; }
    }
    __syncthreads();

    if (tid < CWD) {
        float smy = 0.0f, mmy = -3.0e38f, smx = 0.0f, mmx = -3.0e38f;
#pragma unroll
        for (int w8 = 0; w8 < 8; ++w8) {
            smy += psum[w8][tid];
            mmy = fmaxf(mmy, pmax[w8][tid]);
        }
#pragma unroll 8
        for (int ci = 0; ci < CC; ++ci) {
            float xv = xf[ci * CWD + tid];
            smx += xv;
            mmx = fmaxf(mmx, xv);
        }
        float* fb = feat + (size_t)img * 4 * CHW + (size_t)row * CWD + tid;
        fb[0 * CHW] = smx * (1.0f / CC);   // mean(x)
        fb[1 * CHW] = mmx;                 // max(x)
        fb[2 * CHW] = smy * (1.0f / CC);   // mean(y)
        fb[3 * CHW] = mmy;                 // max(y)
    }
}

// ---------------- Kernel 2: per-(b,c) attention, accumulate w_du[c]*O ----------------
// One block per (b, 4-channel chunk). LDS ~229 KB (one WG per WGP).
__global__ __launch_bounds__(256) void k_attn(
    const float* __restrict__ feat, const float* __restrict__ w_qkv,
    const float* __restrict__ b_qkv, const float* __restrict__ w_du,
    float* __restrict__ du)
{
    __shared__ bf16t Qb[CH * CWD];          // 32 KB, reused as P after softmax
    __shared__ bf16t Kb[CH * CWD];          // 32 KB, row-major [g][w] (k=w contiguous)
    __shared__ bf16t VbT[CWD * BP];         // 34 KB, TRANSPOSED [w][g] (k=g contiguous)
    __shared__ float Sf[CH * SPITCH];       // 66 KB softmax scratch
    __shared__ float accl[CH * CWD];        // 64 KB du-logit partial
    __shared__ float rq[CH], rk[CH];

    const int tid = threadIdx.x;
    const int img = blockIdx.x / (CC / CCHUNK);
    const int c0  = (blockIdx.x % (CC / CCHUNK)) * CCHUNK;

#pragma unroll
    for (int it = 0; it < 64; ++it) accl[tid + it * 256] = 0.0f;

    const int wid  = tid >> 5, lane = tid & 31;
    const int half = lane >> 4, nn = lane & 15;
    const float* fb = feat + (size_t)img * 4 * CHW;

    for (int cc = 0; cc < CCHUNK; ++cc) {
        const int c = c0 + cc;
        const float wq0 = w_qkv[c * 4 + 0], wq1 = w_qkv[c * 4 + 1];
        const float wq2 = w_qkv[c * 4 + 2], wq3 = w_qkv[c * 4 + 3];
        const float wk0 = w_qkv[(CC + c) * 4 + 0], wk1 = w_qkv[(CC + c) * 4 + 1];
        const float wk2 = w_qkv[(CC + c) * 4 + 2], wk3 = w_qkv[(CC + c) * 4 + 3];
        const float wv0 = w_qkv[(2 * CC + c) * 4 + 0], wv1 = w_qkv[(2 * CC + c) * 4 + 1];
        const float wv2 = w_qkv[(2 * CC + c) * 4 + 2], wv3 = w_qkv[(2 * CC + c) * 4 + 3];
        const float bq = b_qkv[c], bk = b_qkv[CC + c], bv = b_qkv[2 * CC + c];
        __syncthreads();   // previous iteration done with Q/K/V before overwrite

        // --- Q,K,V generation + row sum-of-squares (thread -> row tid>>1, 64 cols)
        {
            const int h = tid >> 1;
            const int wbase = (tid & 1) << 6;
            float ssq = 0.0f, ssk = 0.0f;
            const float* f0 = fb + h * CWD;
#pragma unroll 4
            for (int j = 0; j < 64; j += 4) {
                int w = wbase + j;
                float4 a0 = *reinterpret_cast<const float4*>(f0 + w);
                float4 a1 = *reinterpret_cast<const float4*>(f0 + CHW + w);
                float4 a2 = *reinterpret_cast<const float4*>(f0 + 2 * CHW + w);
                float4 a3 = *reinterpret_cast<const float4*>(f0 + 3 * CHW + w);
                v4bf qp, kp;
#pragma unroll
                for (int u = 0; u < 4; ++u) {
                    float b0 = (&a0.x)[u], b1 = (&a1.x)[u];
                    float b2 = (&a2.x)[u], b3 = (&a3.x)[u];
                    float qv = wq0 * b0 + wq1 * b1 + wq2 * b2 + wq3 * b3 + bq;
                    float kv = wk0 * b0 + wk1 * b1 + wk2 * b2 + wk3 * b3 + bk;
                    float vv = wv0 * b0 + wv1 * b1 + wv2 * b2 + wv3 * b3 + bv;
                    qp[u] = (bf16t)qv;
                    kp[u] = (bf16t)kv;
                    VbT[(w + u) * BP + h] = (bf16t)vv;  // transposed: K(=g=h) contiguous
                    ssq += qv * qv;
                    ssk += kv * kv;
                }
                *reinterpret_cast<v4bf*>(&Qb[h * CWD + w]) = qp;
                *reinterpret_cast<v4bf*>(&Kb[h * CWD + w]) = kp;
            }
            ssq += __shfl_xor(ssq, 1);
            ssk += __shfl_xor(ssk, 1);
            if ((tid & 1) == 0) {
                // 1/max(sqrt(ss),1e-12) via v_rsq_f32 with an underflow guard
                rq[h] = (ssq > 1e-24f) ? __builtin_amdgcn_rsqf(ssq) : 1e12f;
                rk[h] = (ssk > 1e-24f) ? __builtin_amdgcn_rsqf(ssk) : 1e12f;
            }
        }
        __syncthreads();

        // --- S = Q K^T  (wave wid owns h-tile wid; loops the 8 g-tiles)
        {
            v16bf Aq[4];
#pragma unroll
            for (int kt = 0; kt < 4; ++kt)
                Aq[kt] = load_A16x32(Qb + (wid * 16) * CWD + kt * 32, CWD);
#pragma unroll
            for (int gt = 0; gt < 8; ++gt) {
                v8f acc = {};
#pragma unroll
                for (int kt = 0; kt < 4; ++kt) {
                    // B[k=w][n=g] = Kb[g*CWD + w]: k contiguous
                    v16bf Bt = load_B_k1(Kb + (gt * 16) * CWD + kt * 32, CWD);
                    acc = __builtin_amdgcn_wmma_f32_16x16x32_bf16(
                        false, Aq[kt], false, Bt, (short)0, acc, false, false);
                }
#pragma unroll
                for (int e = 0; e < 8; ++e) {
                    int h = wid * 16 + half * 8 + e;
                    Sf[h * SPITCH + gt * 16 + nn] = acc[e];
                }
            }
        }
        __syncthreads();

        // --- softmax over g (norm scaling folded in); P (bf16) overwrites Qb
        if (tid < CH) {
            const int h = tid;
            const float sq = rq[h];
            float mx = -3.0e38f;
            for (int g = 0; g < CH; ++g)
                mx = fmaxf(mx, Sf[h * SPITCH + g] * sq * rk[g]);
            float sum = 0.0f;
            for (int g = 0; g < CH; ++g) {
                float e = __expf(Sf[h * SPITCH + g] * sq * rk[g] - mx);
                Sf[h * SPITCH + g] = e;
                sum += e;
            }
            float inv = __builtin_amdgcn_rcpf(sum);
            for (int g = 0; g < CH; ++g)
                Qb[h * CWD + g] = (bf16t)(Sf[h * SPITCH + g] * inv);
        }
        __syncthreads();

        // --- O = P V ; accumulate w_du[c] * O into accl
        {
            const float wd = w_du[c];
            v16bf Ap[4];
#pragma unroll
            for (int kt = 0; kt < 4; ++kt)
                Ap[kt] = load_A16x32(Qb + (wid * 16) * CWD + kt * 32, CWD);
#pragma unroll
            for (int wt = 0; wt < 8; ++wt) {
                v8f acc = {};
#pragma unroll
                for (int kt = 0; kt < 4; ++kt) {
                    // B[k=g][n=w] = VbT[w*BP + g]: k contiguous
                    v16bf Bt = load_B_k1(VbT + (wt * 16) * BP + kt * 32, BP);
                    acc = __builtin_amdgcn_wmma_f32_16x16x32_bf16(
                        false, Ap[kt], false, Bt, (short)0, acc, false, false);
                }
#pragma unroll
                for (int e = 0; e < 8; ++e) {
                    int h = wid * 16 + half * 8 + e;
                    accl[h * CWD + wt * 16 + nn] += wd * acc[e];
                }
            }
        }
    }
    __syncthreads();

    float* dub = du + (size_t)img * CHW;
#pragma unroll
    for (int it = 0; it < 64; ++it) {
        int l = tid + it * 256;
        atomicAdd(&dub[l], accl[l]);
    }
}

// ---------------- Kernel 3: out = x * sigmoid(du), float4 streaming ----------------
__global__ __launch_bounds__(256) void k_final(
    const float* __restrict__ x, const float* __restrict__ du,
    float* __restrict__ out)
{
    size_t i4 = ((size_t)blockIdx.x * 256 + threadIdx.x) * 4;
    const float4 xv = *reinterpret_cast<const float4*>(x + i4);
    size_t b = i4 >> 21;                 // / (C*H*W)
    size_t pix = i4 & (size_t)(CHW - 1);
    const float4 dv = *reinterpret_cast<const float4*>(du + b * CHW + pix);
    float4 o;
    o.x = xv.x * sigmoidf_(dv.x);
    o.y = xv.y * sigmoidf_(dv.y);
    o.z = xv.z * sigmoidf_(dv.z);
    o.w = xv.w * sigmoidf_(dv.w);
    *reinterpret_cast<float4*>(out + i4) = o;
}

extern "C" void kernel_launch(void* const* d_in, const int* in_sizes, int n_in,
                              void* d_out, int out_size, void* d_ws, size_t ws_size,
                              hipStream_t stream) {
    const float* x     = (const float*)d_in[0];
    const float* w_cw  = (const float*)d_in[1];
    const float* b_cw  = (const float*)d_in[2];
    const float* w_qkv = (const float*)d_in[3];
    const float* b_qkv = (const float*)d_in[4];
    const float* w_du  = (const float*)d_in[5];
    const float* b_du  = (const float*)d_in[6];
    float* out  = (float*)d_out;

    float* feat = (float*)d_ws;                       // B*4*H*W = 4 MB
    float* du   = feat + (size_t)CB * 4 * CHW;        // B*H*W   = 1 MB

    k_init_du<<<(CB * CHW + 255) / 256, 256, 0, stream>>>(du, b_du);
    k_gate_feat<<<CB * CH, 256, 0, stream>>>(x, w_cw, b_cw, feat);
    k_attn<<<CB * (CC / CCHUNK), 256, 0, stream>>>(feat, w_qkv, b_qkv, w_du, du);
    k_final<<<(CB * CC * CHW) / (4 * 256), 256, 0, stream>>>(x, du, out);
}
